// Epipolar_23235773071387
// MI455X (gfx1250) — matrix operand order repaired and checked by reading, not compiled, not run on hardware
//
#include <hip/hip_runtime.h>
#include <hip/hip_fp16.h>

typedef __attribute__((ext_vector_type(16))) _Float16 v16h;
typedef __attribute__((ext_vector_type(8)))  float    v8f;

namespace {

constexpr int NL  = 4096;   // query tokens (T*H*W)
constexpr int NC  = 320;    // channels
constexpr int NH  = 8;      // heads
constexpr int ND  = 40;     // head dim
constexpr int NKtrue = 4097;// context length (1 reg token + NL)
constexpr int NKP = 4128;   // context padded to multiple of 32 (32*129)
constexpr int NDV = 48;     // head dim padded to 3x16 for PV tiles
constexpr int NTP = 10;     // 10 column-tile PAIRS in a 320-wide GEMM

union U16 { v16h v; _Float16 h[16]; unsigned int u[8]; };

// A-operand (16-bit 16x32) K offset for VGPR v, lane-half: ISA 7.12.2
__device__ __forceinline__ int koffA(int v, int half) {
  return half * 8 + ((v < 4) ? 2 * v : 8 + 2 * v);
}
// B-operand (32x16, 16-bit) K offset: same K order as the A operand
__device__ __forceinline__ int koffB(int v, int half) {
  return half * 8 + ((v < 4) ? 2 * v : 8 + 2 * v);
}

__device__ __forceinline__ v8f wmma16(v16h a, v16h b, v8f c) {
  return __builtin_amdgcn_wmma_f32_16x16x32_f16(
      false, a, false, b, (short)0, c, false, false);
}

// contiguous K-pair B-operand load (source laid out with K contiguous per n)
__device__ __forceinline__ v16h load_b_pairs(const _Float16* base, int half) {
  U16 b;
#pragma unroll
  for (int v = 0; v < 8; ++v)
    b.u[v] = *reinterpret_cast<const unsigned int*>(base + koffB(v, half));
  return b.v;
}

// ---------------- prep kernels ----------------

// features (1,16,320,16,16) -> x16[L][C] f16, and ctx16 rows 1..4096
__global__ void k_build_x(const float* __restrict__ feat,
                          _Float16* __restrict__ x16,
                          _Float16* __restrict__ ctx16) {
  int idx = blockIdx.x * 256 + threadIdx.x;
  if (idx >= NL * NC) return;
  int l = idx / NC, c = idx % NC;
  int t = l >> 8, hw = l & 255, hh = hw >> 4, ww = hw & 15;
  float v = feat[(((size_t)t * NC + c) * 16 + hh) * 16 + ww];
  _Float16 h = (_Float16)v;
  x16[(size_t)l * NC + c] = h;
  ctx16[(size_t)(l + 1) * NC + c] = h;
}

// ctx row 0 = reg token; rows 4097..4127 = 0
__global__ void k_ctx_misc(const float* __restrict__ regtok,
                           _Float16* __restrict__ ctx16) {
  int idx = blockIdx.x * 256 + threadIdx.x;
  if (idx < NC) {
    ctx16[idx] = (_Float16)regtok[idx];
  } else {
    int j = idx - NC;
    if (j >= (NKP - NKtrue) * NC) return;
    ctx16[(size_t)NKtrue * NC + j] = (_Float16)0.f;
  }
}

// convert f32 weight [NC][NC] -> TRANSPOSED f16 weight Wt[n][k]
__global__ void k_cvt_t(const float* __restrict__ src,
                        _Float16* __restrict__ dst) {
  int idx = blockIdx.x * 256 + threadIdx.x;
  if (idx >= NC * NC) return;
  int k = idx / NC, n = idx % NC;
  dst[(size_t)n * NC + k] = (_Float16)src[idx];
}

__global__ void k_zero16(_Float16* __restrict__ dst, int n) {
  int idx = blockIdx.x * 256 + threadIdx.x;
  if (idx < n) dst[idx] = (_Float16)0.f;
}

// ------- WMMA GEMM: out[M x 320] = A[M x 320] * Bt^T (Bt = [320n][320k]) ----
// Each wave owns a 16x32 output tile (2 N-tiles share one A operand).
// mode 0: store f16 row-major (ld=NC). mode 1: store transposed-per-head into
// vT[h][d][m] with vT row stride NKP (h = n/40, d = n%40).
__global__ __launch_bounds__(256) void k_gemm(
    const _Float16* __restrict__ A, int M,
    const _Float16* __restrict__ Bt,
    _Float16* __restrict__ out, int mode) {
  int lane = threadIdx.x, half = lane >> 4, l16 = lane & 15;
  int tile = blockIdx.x * 8 + threadIdx.y;
  if (tile >= ((M + 15) / 16) * NTP) return;
  int tm = tile / NTP, tn = tile % NTP;

  v8f acc0 = {}, acc1 = {};
  const _Float16* arow = A + (size_t)(tm * 16 + l16) * NC;
  int col0 = tn * 32 + l16, col1 = col0 + 16;
  const _Float16* b0row = Bt + (size_t)col0 * NC;
  const _Float16* b1row = Bt + (size_t)col1 * NC;
#pragma unroll 2
  for (int k0 = 0; k0 < NC; k0 += 32) {
    U16 a;
#pragma unroll
    for (int v = 0; v < 8; ++v)
      a.u[v] = *reinterpret_cast<const unsigned int*>(arow + k0 + koffA(v, half));
    acc0 = wmma16(a.v, load_b_pairs(b0row + k0, half), acc0);
    acc1 = wmma16(a.v, load_b_pairs(b1row + k0, half), acc1);
  }
  if (mode == 0) {
#pragma unroll
    for (int r = 0; r < 8; ++r) {
      size_t row = (size_t)(tm * 16 + r + 8 * half) * NC;
      out[row + col0] = (_Float16)acc0[r];
      out[row + col1] = (_Float16)acc1[r];
    }
  } else {
    int h0 = col0 / ND, d0 = col0 % ND;
    int h1 = col1 / ND, d1 = col1 % ND;
#pragma unroll
    for (int r = 0; r < 8; ++r) {
      int m = tm * 16 + r + 8 * half;
      out[(size_t)(h0 * NDV + d0) * NKP + m] = (_Float16)acc0[r];
      out[(size_t)(h1 * NDV + d1) * NKP + m] = (_Float16)acc1[r];
    }
  }
}

// ---------------- flash attention ----------------
// grid = 256 q-tiles, block (32,8): wave = head. Mask tile shared in LDS.
__global__ __launch_bounds__(256) void k_flash(
    const _Float16* __restrict__ q16,   // [NL][NC]
    const _Float16* __restrict__ k16,   // [NKP][NC]
    const _Float16* __restrict__ vT,    // [NH][NDV][NKP]
    const unsigned char* __restrict__ mask,  // [NL][NL] bool
    _Float16* __restrict__ attn16) {    // [NL][NC]
  __shared__ unsigned char smask[16 * 32];
  __shared__ _Float16 sP[NH][16][32];

  const int h = threadIdx.y, lane = threadIdx.x;
  const int half = lane >> 4, l16 = lane & 15;
  const int qb = blockIdx.x, qrow = qb * 16;
  const int tid = h * 32 + lane;
  const float sc = 0.15811388300841898f;  // 40^-0.5

  // Q A-operands: head-dim padded 40 -> 64 (two k=32 chunks)
  const _Float16* qbase = q16 + (size_t)(qrow + l16) * NC + h * ND;
  v16h a0, a1;
  {
    U16 a;
#pragma unroll
    for (int v = 0; v < 8; ++v)
      a.u[v] = *reinterpret_cast<const unsigned int*>(qbase + koffA(v, half));
    a0 = a.v;
#pragma unroll
    for (int v = 0; v < 8; ++v) {
      int kk = 32 + koffA(v, half);
      a.u[v] = (kk < ND)
          ? *reinterpret_cast<const unsigned int*>(qbase + kk) : 0u;
    }
    a1 = a.v;
  }

  float mrow[8], lrow[8];
  v8f o[3];
#pragma unroll
  for (int r = 0; r < 8; ++r) { mrow[r] = -3.0e38f; lrow[r] = 0.f; }
#pragma unroll
  for (int t = 0; t < 3; ++t) o[t] = v8f{};

  for (int kc = 0; kc < NKP; kc += 32) {
    __syncthreads();
    // cooperative mask tile: 16 rows x 32 padded-context cols
    for (int e = tid; e < 512; e += 256) {
      int m = e >> 5, jj = e & 31, j = kc + jj;
      unsigned char mv;
      if (j == 0) mv = 1;                               // register token
      else if (j <= NL) mv = mask[(size_t)(qrow + m) * NL + (j - 1)];
      else mv = 0;                                      // pad columns
      smask[e] = mv;
    }
    __syncthreads();

    // S = Q K^T for 2 key subtiles (cols kc..kc+15, kc+16..kc+31)
    v8f s0 = {}, s1 = {};
    {
      const _Float16* kb0 = k16 + (size_t)(kc + l16) * NC + h * ND;
      const _Float16* kb1 = k16 + (size_t)(kc + 16 + l16) * NC + h * ND;
      if (kc + 32 < NKP)  // prefetch next chunk's K rows into L1/L2
        __builtin_prefetch(kb0 + (size_t)32 * NC, 0, 3);
      U16 b;
      s0 = wmma16(a0, load_b_pairs(kb0, half), s0);
#pragma unroll
      for (int v = 0; v < 8; ++v) {
        int kk = koffB(v, half);
        b.u[v] = (32 + kk < ND)
            ? *reinterpret_cast<const unsigned int*>(kb0 + 32 + kk) : 0u;
      }
      s0 = wmma16(a1, b.v, s0);
      s1 = wmma16(a0, load_b_pairs(kb1, half), s1);
#pragma unroll
      for (int v = 0; v < 8; ++v) {
        int kk = koffB(v, half);
        b.u[v] = (32 + kk < ND)
            ? *reinterpret_cast<const unsigned int*>(kb1 + 32 + kk) : 0u;
      }
      s1 = wmma16(a1, b.v, s1);
    }

    // scale + mask + online softmax (rows live across 16-lane groups)
    float p0[8], p1[8], alpha[8], rsum[8], rmax[8];
#pragma unroll
    for (int r = 0; r < 8; ++r) {
      int m = r + 8 * half;
      float v0 = smask[m * 32 + l16]      ? s0[r] * sc : -3.0e38f;
      float v1 = smask[m * 32 + 16 + l16] ? s1[r] * sc : -3.0e38f;
      p0[r] = v0; p1[r] = v1;
      rmax[r] = fmaxf(v0, v1);
    }
#pragma unroll
    for (int off = 1; off < 16; off <<= 1)
#pragma unroll
      for (int r = 0; r < 8; ++r)
        rmax[r] = fmaxf(rmax[r], __shfl_xor(rmax[r], off, 16));
#pragma unroll
    for (int r = 0; r < 8; ++r) {
      float mnew = fmaxf(mrow[r], rmax[r]);
      alpha[r] = __expf(mrow[r] - mnew);
      mrow[r] = mnew;
      p0[r] = __expf(p0[r] - mnew);
      p1[r] = __expf(p1[r] - mnew);
      rsum[r] = p0[r] + p1[r];
    }
#pragma unroll
    for (int off = 1; off < 16; off <<= 1)
#pragma unroll
      for (int r = 0; r < 8; ++r)
        rsum[r] += __shfl_xor(rsum[r], off, 16);
#pragma unroll
    for (int r = 0; r < 8; ++r)
      lrow[r] = lrow[r] * alpha[r] + rsum[r];

    // stage P (C-layout -> LDS) then reload in A-layout (wave-local)
#pragma unroll
    for (int r = 0; r < 8; ++r) {
      int m = r + 8 * half;
      sP[h][m][l16]      = (_Float16)p0[r];
      sP[h][m][16 + l16] = (_Float16)p1[r];
    }
    v16h ap;
    {
      U16 a;
      const unsigned int* prow =
          reinterpret_cast<const unsigned int*>(&sP[h][l16][0]);
#pragma unroll
      for (int v = 0; v < 8; ++v) a.u[v] = prow[koffA(v, half) >> 1];
      ap = a.v;
    }

    // O = O*alpha + P V  (3 head-dim tiles of 16)
#pragma unroll
    for (int t = 0; t < 3; ++t) {
#pragma unroll
      for (int r = 0; r < 8; ++r) o[t][r] *= alpha[r];
      const _Float16* vb =
          vT + (size_t)(h * NDV + 16 * t + l16) * NKP + kc;
      o[t] = wmma16(ap, load_b_pairs(vb, half), o[t]);
    }
  }

  // normalize and store attention output (f16), head-dim 0..39 only
#pragma unroll
  for (int r = 0; r < 8; ++r) {
    float inv = 1.f / lrow[r];
    int row = qrow + r + 8 * half;
    _Float16* dst = attn16 + (size_t)row * NC + h * ND;
    dst[l16]      = (_Float16)(o[0][r] * inv);
    dst[16 + l16] = (_Float16)(o[1][r] * inv);
    if (l16 < 8) dst[32 + l16] = (_Float16)(o[2][r] * inv);
  }
}

// ----- epilogue: y = attn * Wo + bo, fused output permutation (hw,t,c) ------
__global__ __launch_bounds__(256) void k_out(
    const _Float16* __restrict__ A,    // attn16 [NL][NC]
    const _Float16* __restrict__ Bt,   // wo16t [320n][320k]
    const float* __restrict__ bo,
    float* __restrict__ out) {
  int lane = threadIdx.x, half = lane >> 4, l16 = lane & 15;
  int tile = blockIdx.x * 8 + threadIdx.y;
  if (tile >= (NL / 16) * NTP) return;
  int tm = tile / NTP, tn = tile % NTP;

  v8f acc0 = {}, acc1 = {};
  const _Float16* arow = A + (size_t)(tm * 16 + l16) * NC;
  int col0 = tn * 32 + l16, col1 = col0 + 16;
  const _Float16* b0row = Bt + (size_t)col0 * NC;
  const _Float16* b1row = Bt + (size_t)col1 * NC;
#pragma unroll 2
  for (int k0 = 0; k0 < NC; k0 += 32) {
    U16 a;
#pragma unroll
    for (int v = 0; v < 8; ++v)
      a.u[v] = *reinterpret_cast<const unsigned int*>(arow + k0 + koffA(v, half));
    acc0 = wmma16(a.v, load_b_pairs(b0row + k0, half), acc0);
    acc1 = wmma16(a.v, load_b_pairs(b1row + k0, half), acc1);
  }
  float bias0 = bo[col0], bias1 = bo[col1];
#pragma unroll
  for (int r = 0; r < 8; ++r) {
    int row = tm * 16 + r + 8 * half;                 // l in [0,4096)
    int dst = ((row & 255) * 16 + (row >> 8)) * NC;   // (hw, t, c)
    out[dst + col0] = acc0[r] + bias0;
    out[dst + col1] = acc1[r] + bias1;
  }
}

}  // namespace

extern "C" void kernel_launch(void* const* d_in, const int* in_sizes, int n_in,
                              void* d_out, int out_size, void* d_ws,
                              size_t ws_size, hipStream_t stream) {
  (void)in_sizes; (void)n_in; (void)out_size; (void)ws_size;
  const float* feat          = (const float*)d_in[0];
  const unsigned char* msk   = (const unsigned char*)d_in[1];  // jnp bool = 1B
  const float* Wq            = (const float*)d_in[2];
  const float* Wk            = (const float*)d_in[3];
  const float* Wv            = (const float*)d_in[4];
  const float* regtok        = (const float*)d_in[5];
  const float* Wo            = (const float*)d_in[6];
  const float* bo            = (const float*)d_in[7];
  float* out                 = (float*)d_out;

  _Float16* ws = (_Float16*)d_ws;
  size_t off = 0;
  _Float16* x16   = ws + off; off += (size_t)NL * NC;
  _Float16* ctx16 = ws + off; off += (size_t)NKP * NC;
  _Float16* wq16t = ws + off; off += (size_t)NC * NC;
  _Float16* wk16t = ws + off; off += (size_t)NC * NC;
  _Float16* wv16t = ws + off; off += (size_t)NC * NC;
  _Float16* wo16t = ws + off; off += (size_t)NC * NC;
  _Float16* q16   = ws + off; off += (size_t)NL * NC;
  _Float16* k16   = ws + off; off += (size_t)NKP * NC;
  _Float16* vT    = ws + off; off += (size_t)NH * NDV * NKP;
  _Float16* att16 = ws + off; off += (size_t)NL * NC;
  // ~17.2 MB of f16 workspace total

  dim3 wb(32, 8);
  int nvt = NH * NDV * NKP;
  k_zero16<<<(nvt + 255) / 256, 256, 0, stream>>>(vT, nvt);
  k_build_x<<<(NL * NC) / 256, 256, 0, stream>>>(feat, x16, ctx16);
  k_ctx_misc<<<(NC + (NKP - NKtrue) * NC + 255) / 256, 256, 0, stream>>>(
      regtok, ctx16);
  k_cvt_t<<<(NC * NC) / 256, 256, 0, stream>>>(Wq, wq16t);
  k_cvt_t<<<(NC * NC) / 256, 256, 0, stream>>>(Wk, wk16t);
  k_cvt_t<<<(NC * NC) / 256, 256, 0, stream>>>(Wv, wv16t);
  k_cvt_t<<<(NC * NC) / 256, 256, 0, stream>>>(Wo, wo16t);

  int tq = (NL / 16) * NTP;          // 2560 tiles
  int tk = (NKP / 16) * NTP;         // 2580 tiles
  k_gemm<<<(tq + 7) / 8, wb, 0, stream>>>(x16, NL, wq16t, q16, 0);
  k_gemm<<<(tk + 7) / 8, wb, 0, stream>>>(ctx16, NKP, wk16t, k16, 0);
  k_gemm<<<(tk + 7) / 8, wb, 0, stream>>>(ctx16, NKP, wv16t, vT, 1);

  k_flash<<<NL / 16, wb, 0, stream>>>(q16, k16, vT, msk, att16);
  k_out<<<(tq + 7) / 8, wb, 0, stream>>>(att16, wo16t, bo, out);
}
